// LRUEmbedding_52484500357265
// MI455X (gfx1250) — matrix operand order repaired and checked by reading, not compiled, not run on hardware
//
#include <hip/hip_runtime.h>
#include <hip/hip_bf16.h>
#include <stdint.h>
#include <stddef.h>

// ---------------------------------------------------------------------------
// LRU (3 stacked layers, diagonal complex SSM + GLU) for MI455X / gfx1250.
// Strategy:
//   * Stage 1 per layer: fused bf16 WMMA GEMM  u @ [gamma*B_re; gamma*B_im; D]^T
//   * Stage 2: chunked complex scan (3 phases) in f32
//   * Stage 3: fused bf16 WMMA GEMM [st_re|st_im] @ [C_re|-C_im]^T + Du, GLU
//   * Final: mean over T
// ---------------------------------------------------------------------------

typedef __attribute__((ext_vector_type(16))) __bf16        v16bf;
typedef __attribute__((ext_vector_type(8)))  float         v8f;
typedef __attribute__((ext_vector_type(4)))  unsigned int  v4u;

union FragAB {
  v4u   q[2];
  v16bf v;
};

__device__ __forceinline__ unsigned short f2bf(float f) {
  union { float f; unsigned u; } c; c.f = f;
  unsigned u = c.u;
  u += 0x7fffu + ((u >> 16) & 1u);   // round-to-nearest-even
  return (unsigned short)(u >> 16);
}
__device__ __forceinline__ float bf2f(unsigned short h) {
  union { unsigned u; float f; } c; c.u = ((unsigned)h) << 16;
  return c.f;
}

struct C2 { float re, im; };

#define B_SZ    16
#define T_SEQ   4096
#define M_ROWS  (B_SZ * T_SEQ)   // 65536
#define NSTATE  256
#define CHUNK   128
#define NCHUNK  (T_SEQ / CHUNK)  // 32

// ---------------------------------------------------------------------------
// Weight / lambda preparation (per layer, tiny).
// Wa rows: [0,256)=gamma*B_re, [256,512)=gamma*B_im, [512,512+2*dout)=D  (x din)
// Wc rows: 2*dout x 512 : [k<256]=C_re[o,k], [k>=256]=-C_im[o,k-256]
// lam: [0..255]=re, [256..511]=im
// ---------------------------------------------------------------------------
__global__ __launch_bounds__(256) void k_prep_layer(
    const float* __restrict__ nu_log, const float* __restrict__ theta_log,
    const float* __restrict__ gamma_log,
    const float* __restrict__ Bre, const float* __restrict__ Bim,
    const float* __restrict__ Cre, const float* __restrict__ Cim,
    const float* __restrict__ Dm,
    unsigned short* __restrict__ Wa, unsigned short* __restrict__ Wc,
    float* __restrict__ lam, int din, int dout)
{
  const int idx    = blockIdx.x * 256 + threadIdx.x;
  const int stride = gridDim.x * 256;
  if (idx < NSTATE) {
    const float r = __expf(-__expf(nu_log[idx]));
    const float t = __expf(theta_log[idx]);
    lam[idx]          = r * cosf(t);
    lam[NSTATE + idx] = r * sinf(t);
  }
  const int rowsA  = 512 + 2 * dout;
  const int totalA = rowsA * din;
  for (int i = idx; i < totalA; i += stride) {
    const int r = i / din, c = i - r * din;
    float v;
    if (r < 256)       v = __expf(gamma_log[r])       * Bre[r * din + c];
    else if (r < 512)  v = __expf(gamma_log[r - 256]) * Bim[(r - 256) * din + c];
    else               v = Dm[(r - 512) * din + c];
    Wa[i] = f2bf(v);
  }
  const int totalC = (2 * dout) * 512;
  for (int i = idx; i < totalC; i += stride) {
    const int o = i >> 9, k = i & 511;
    const float v = (k < 256) ? Cre[o * 256 + k] : -Cim[o * 256 + (k - 256)];
    Wc[i] = f2bf(v);
  }
}

__global__ __launch_bounds__(256) void k_to_bf16(
    const float* __restrict__ in, unsigned short* __restrict__ out, size_t n)
{
  size_t i = (size_t)blockIdx.x * 256 + threadIdx.x;
  if (i < n) out[i] = f2bf(in[i]);
}

// ---------------------------------------------------------------------------
// GEMM 1: Out(f32, M x Nout) = A(bf16, M x K) @ W(bf16, Nout x K)^T
// Block: 256 thr = 8 waves (2m x 4n), tile 64 x 256; wave tile 32 x 64.
// Fragments loaded in contiguous k order (lane half h holds k = 16h..16h+15), two b128 per lane each.
// ---------------------------------------------------------------------------
__global__ __launch_bounds__(256) void k_gemm_in(
    const unsigned short* __restrict__ A,
    const unsigned short* __restrict__ W,
    float* __restrict__ Out,
    int K, int Nout)
{
  const int lane = threadIdx.x & 31;
  const int wid  = threadIdx.x >> 5;
  const int wm   = wid & 1;
  const int wn   = wid >> 1;
  const int lrow = lane & 15;
  const int lhi  = lane >> 4;
  const int row0 = blockIdx.x * 64 + wm * 32;
  const int col0 = blockIdx.y * 256 + wn * 64;

  v8f acc[2][4] = {};

  for (int k0 = 0; k0 < K; k0 += 32) {
    FragAB a[2];
#pragma unroll
    for (int i = 0; i < 2; ++i) {
      // A 16x32 bf16 frag: lanes 0-15 -> K kb..kb+15; lanes 16-31 -> +16
      const unsigned short* p = A + (size_t)(row0 + i * 16 + lrow) * K + (k0 + lhi * 16);
      a[i].q[0] = *(const v4u*)p;
      a[i].q[1] = *(const v4u*)(p + 8);
    }
#pragma unroll
    for (int j = 0; j < 4; ++j) {
      // B 32x16 bf16 frag: lane group selects K-half; B[k,n] = W[n,k]
      const unsigned short* p = W + (size_t)(col0 + j * 16 + lrow) * K + (k0 + lhi * 16);
      FragAB b;
      b.q[0] = *(const v4u*)p;
      b.q[1] = *(const v4u*)(p + 8);
#pragma unroll
      for (int i = 0; i < 2; ++i)
        acc[i][j] = __builtin_amdgcn_wmma_f32_16x16x32_bf16(
            false, a[i].v, false, b.v, (short)0, acc[i][j], false, false);
    }
  }

#pragma unroll
  for (int i = 0; i < 2; ++i)
#pragma unroll
    for (int j = 0; j < 4; ++j) {
      const int c = col0 + j * 16 + lrow;
#pragma unroll
      for (int r = 0; r < 8; ++r) {
        const int rr = row0 + i * 16 + lhi * 8 + r;   // C/D 16x16 f32 layout
        Out[(size_t)rr * Nout + c] = acc[i][j][r];
      }
    }
}

// ---------------------------------------------------------------------------
// GEMM 2 + GLU: y = S(bf16, M x 512) @ Wc(2*dout x 512)^T + Du
//               u_next[:, c] = (y_h) * sigmoid(y_g),  g = column c + dout
// Block: 128 thr = 4 waves (2m x 2n), tile 64 x 128 output columns.
// Each wave accumulates BOTH h and g tiles (2m x 4n x 2 WMMA per K-step).
// ---------------------------------------------------------------------------
__global__ __launch_bounds__(128) void k_gemm_out_glu(
    const unsigned short* __restrict__ S,
    const unsigned short* __restrict__ Wc,
    const float* __restrict__ BiasBuf, int ldb,   // Du lives at columns [512, 512+2*dout)
    unsigned short* __restrict__ Unext,
    int dout)
{
  const int lane = threadIdx.x & 31;
  const int wid  = threadIdx.x >> 5;   // 0..3
  const int wm   = wid & 1;
  const int wn   = wid >> 1;
  const int lrow = lane & 15;
  const int lhi  = lane >> 4;
  const int row0 = blockIdx.x * 64 + wm * 32;
  const int col0 = blockIdx.y * 128 + wn * 64;
  const int K    = 512;

  v8f accH[2][4] = {};
  v8f accG[2][4] = {};

  for (int k0 = 0; k0 < K; k0 += 32) {
    FragAB a[2];
#pragma unroll
    for (int i = 0; i < 2; ++i) {
      const unsigned short* p = S + (size_t)(row0 + i * 16 + lrow) * K + (k0 + lhi * 16);
      a[i].q[0] = *(const v4u*)p;
      a[i].q[1] = *(const v4u*)(p + 8);
    }
#pragma unroll
    for (int j = 0; j < 4; ++j) {
      const int ch = col0 + j * 16 + lrow;
      const unsigned short* ph = Wc + (size_t)ch * K + (k0 + lhi * 16);
      const unsigned short* pg = Wc + (size_t)(ch + dout) * K + (k0 + lhi * 16);
      FragAB bh, bg;
      bh.q[0] = *(const v4u*)ph;  bh.q[1] = *(const v4u*)(ph + 8);
      bg.q[0] = *(const v4u*)pg;  bg.q[1] = *(const v4u*)(pg + 8);
#pragma unroll
      for (int i = 0; i < 2; ++i) {
        accH[i][j] = __builtin_amdgcn_wmma_f32_16x16x32_bf16(
            false, a[i].v, false, bh.v, (short)0, accH[i][j], false, false);
        accG[i][j] = __builtin_amdgcn_wmma_f32_16x16x32_bf16(
            false, a[i].v, false, bg.v, (short)0, accG[i][j], false, false);
      }
    }
  }

#pragma unroll
  for (int i = 0; i < 2; ++i)
#pragma unroll
    for (int j = 0; j < 4; ++j) {
      const int c = col0 + j * 16 + lrow;
#pragma unroll
      for (int r = 0; r < 8; ++r) {
        const int rr = row0 + i * 16 + lhi * 8 + r;
        const float h = accH[i][j][r] + BiasBuf[(size_t)rr * ldb + 512 + c];
        const float g = accG[i][j][r] + BiasBuf[(size_t)rr * ldb + 512 + dout + c];
        const float sg = 1.0f / (1.0f + __expf(-g));
        Unext[(size_t)rr * dout + c] = f2bf(h * sg);
      }
    }
}

// ---------------------------------------------------------------------------
// Chunked complex scan: x_t = lam * x_{t-1} + bu_t  (per channel (b,n))
// Phase 1 (writeStates=0): local scan from 0, emit chunk-end state.
// Phase 3 (writeStates=1): rescan seeded with carry, emit bf16 [re|im] states.
// Grid: (NCHUNK, B_SZ) x 256 threads (thread = state channel n) -> coalesced.
// ---------------------------------------------------------------------------
__global__ __launch_bounds__(256) void k_scan_chunk(
    const float* __restrict__ BU, int ld,
    const float* __restrict__ lam,
    const C2* __restrict__ carryIn,
    C2* __restrict__ chunkEnd,
    unsigned short* __restrict__ Sout,
    int writeStates)
{
  const int n     = threadIdx.x;
  const int chunk = blockIdx.x;
  const int b     = blockIdx.y;
  const int ch    = b * NSTATE + n;
  const float lr = lam[n], li = lam[NSTATE + n];
  float xr = 0.f, xi = 0.f;
  if (writeStates) {
    C2 c0 = carryIn[(size_t)ch * NCHUNK + chunk];
    xr = c0.re; xi = c0.im;
  }
  const size_t rowBase = (size_t)b * T_SEQ + (size_t)chunk * CHUNK;
  for (int j = 0; j < CHUNK; ++j) {
    const size_t row = rowBase + j;
    const float br = BU[row * ld + n];
    const float bi = BU[row * ld + NSTATE + n];
    const float nr = lr * xr - li * xi + br;
    const float ni = lr * xi + li * xr + bi;
    xr = nr; xi = ni;
    if (writeStates) {
      Sout[row * 512 + n]          = f2bf(xr);
      Sout[row * 512 + NSTATE + n] = f2bf(xi);
    }
  }
  if (!writeStates) chunkEnd[(size_t)ch * NCHUNK + chunk] = C2{xr, xi};
}

// Phase 2: sequential combine of chunk carries per channel. lam^CHUNK computed
// exactly from nu_log/theta_log: |lam|^L = exp(-L*exp(nu)), arg = L*exp(theta).
__global__ __launch_bounds__(256) void k_scan_carry(
    const C2* __restrict__ chunkEnd,
    C2* __restrict__ carry,
    const float* __restrict__ nu_log,
    const float* __restrict__ theta_log)
{
  const int ch = blockIdx.x * 256 + threadIdx.x;   // 0..4095
  const int n  = ch & (NSTATE - 1);
  const float mag = __expf(-(float)CHUNK * __expf(nu_log[n]));
  const float ang = (float)CHUNK * __expf(theta_log[n]);
  const float lr = mag * cosf(ang);
  const float li = mag * sinf(ang);
  float cr = 0.f, ci = 0.f;
  for (int c = 0; c < NCHUNK; ++c) {
    carry[(size_t)ch * NCHUNK + c] = C2{cr, ci};
    C2 e = chunkEnd[(size_t)ch * NCHUNK + c];
    const float nr = lr * cr - li * ci + e.re;
    const float ni = lr * ci + li * cr + e.im;
    cr = nr; ci = ni;
  }
}

// Final: out[b,o] = mean_t u[b,t,o]  (u is bf16, 128 wide). 128 threads = o.
__global__ __launch_bounds__(128) void k_reduce_mean(
    const unsigned short* __restrict__ U, float* __restrict__ out)
{
  const int b = blockIdx.x;
  const int o = threadIdx.x;
  float s = 0.f;
  for (int t = 0; t < T_SEQ; ++t)
    s += bf2f(U[((size_t)b * T_SEQ + t) * 128 + o]);
  out[b * 128 + o] = s * (1.0f / (float)T_SEQ);
}

// ---------------------------------------------------------------------------
extern "C" void kernel_launch(void* const* d_in, const int* in_sizes, int n_in,
                              void* d_out, int out_size, void* d_ws, size_t ws_size,
                              hipStream_t stream) {
  (void)in_sizes; (void)n_in; (void)out_size; (void)ws_size;

  const float* x = (const float*)d_in[0];

  // Workspace carve-up (256B aligned regions).
  char* ws = (char*)d_ws;
  size_t off = 0;
  auto carve = [&](size_t bytes) -> char* {
    char* p = ws + off;
    off += (bytes + 255) & ~(size_t)255;
    return p;
  };
  unsigned short* uA     = (unsigned short*)carve((size_t)M_ROWS * 256 * 2);
  unsigned short* uB     = (unsigned short*)carve((size_t)M_ROWS * 256 * 2);
  float*          gout   = (float*)         carve((size_t)M_ROWS * 1024 * 4);
  unsigned short* states = (unsigned short*)carve((size_t)M_ROWS * 512 * 2);
  C2*             chEnd  = (C2*)            carve((size_t)B_SZ * NSTATE * NCHUNK * sizeof(C2));
  C2*             carry  = (C2*)            carve((size_t)B_SZ * NSTATE * NCHUNK * sizeof(C2));
  unsigned short* Wa     = (unsigned short*)carve((size_t)1024 * 256 * 2);
  unsigned short* Wc     = (unsigned short*)carve((size_t)512 * 512 * 2);
  float*          lam    = (float*)         carve((size_t)512 * 4);

  const int dinL[3]  = {64, 256, 256};
  const int doutL[3] = {256, 256, 128};

  // x -> bf16
  {
    const size_t n = (size_t)M_ROWS * 64;
    k_to_bf16<<<dim3((unsigned)((n + 255) / 256)), dim3(256), 0, stream>>>(x, uA, n);
  }

  unsigned short* cur = uA;
  unsigned short* nxt = uB;

  for (int l = 0; l < 3; ++l) {
    const int base = 1 + l * 8;
    const float* nu_log    = (const float*)d_in[base + 0];
    const float* theta_log = (const float*)d_in[base + 1];
    const float* gamma_log = (const float*)d_in[base + 2];
    const float* B_re      = (const float*)d_in[base + 3];
    const float* B_im      = (const float*)d_in[base + 4];
    const float* C_re      = (const float*)d_in[base + 5];
    const float* C_im      = (const float*)d_in[base + 6];
    const float* Dmat      = (const float*)d_in[base + 7];

    const int din   = dinL[l];
    const int dout  = doutL[l];
    const int NoutA = 512 + 2 * dout;   // [bu_re | bu_im | Du]

    k_prep_layer<<<dim3(256), dim3(256), 0, stream>>>(
        nu_log, theta_log, gamma_log, B_re, B_im, C_re, C_im, Dmat,
        Wa, Wc, lam, din, dout);

    k_gemm_in<<<dim3(M_ROWS / 64, NoutA / 256), dim3(256), 0, stream>>>(
        cur, Wa, gout, din, NoutA);

    k_scan_chunk<<<dim3(NCHUNK, B_SZ), dim3(256), 0, stream>>>(
        gout, NoutA, lam, carry, chEnd, states, 0);

    k_scan_carry<<<dim3(B_SZ * NSTATE / 256), dim3(256), 0, stream>>>(
        chEnd, carry, nu_log, theta_log);

    k_scan_chunk<<<dim3(NCHUNK, B_SZ), dim3(256), 0, stream>>>(
        gout, NoutA, lam, carry, chEnd, states, 1);

    k_gemm_out_glu<<<dim3(M_ROWS / 64, dout / 128), dim3(128), 0, stream>>>(
        states, Wc, gout, NoutA, nxt, dout);

    unsigned short* t = cur; cur = nxt; nxt = t;
  }

  k_reduce_mean<<<dim3(B_SZ), dim3(128), 0, stream>>>(cur, (float*)d_out);
}
